// MultiHeadAttention_49632642072952
// MI455X (gfx1250) — matrix operand order repaired and checked by reading, not compiled, not run on hardware
//
#include <hip/hip_runtime.h>
#include <hip/hip_bf16.h>

#define DEV __device__ __forceinline__

typedef __attribute__((ext_vector_type(16))) __bf16 v16bf;
typedef __attribute__((ext_vector_type(8)))  __bf16 v8bf;
typedef __attribute__((ext_vector_type(4)))  __bf16 v4bf;
typedef __attribute__((ext_vector_type(8)))  float  v8f;

constexpr int Bc = 4, Nc = 2048, Dc = 1024, Hc = 16, DHc = 64, Oc = 1024;
constexpr int BNc = Bc * Nc;                 // 8192

// --- CDNA5 async global->LDS copy (ASYNCcnt path), guarded ------------------
#if defined(__has_builtin)
#  if __has_builtin(__builtin_amdgcn_global_load_async_to_lds_b128)
#    define USE_ASYNC_LDS 1
#  endif
#endif
#ifndef USE_ASYNC_LDS
#  define USE_ASYNC_LDS 0
#endif

#define AS1 __attribute__((address_space(1)))
#define AS3 __attribute__((address_space(3)))
typedef int vsi4 __attribute__((vector_size(4 * sizeof(int))));  // matches builtin param

DEV void async_copy16(const __bf16* gp, __bf16* lp) {
#if USE_ASYNC_LDS
  __builtin_amdgcn_global_load_async_to_lds_b128((AS1 vsi4*)gp, (AS3 vsi4*)lp,
                                                 0, 0);
#else
  *(v8bf*)lp = *(const v8bf*)gp;
#endif
}

DEV void wait_async0() {
#if defined(__has_builtin)
#  if __has_builtin(__builtin_amdgcn_s_wait_asynccnt)
  __builtin_amdgcn_s_wait_asynccnt(0);
#  else
  asm volatile("s_wait_asynccnt 0x0" ::: "memory");
#  endif
#else
  asm volatile("s_wait_asynccnt 0x0" ::: "memory");
#endif
}

// ---------------------------------------------------------------------------
// WMMA fragment helpers (ISA 7.12.2 layouts, wave32)
// A 16x32 bf16: lane -> M = lane&15 ; K-chunks [off..off+7] and [off+16..off+23],
//               off = (lane>>4)*8. Two 16B contiguous loads per lane.
// B 32x16 bf16: lane -> N = lane&15 ; same K-chunks as A, koff=(lane>>4)*8.
// C/D 16x16 f32: lane -> N = lane&15 ; VGPR j -> M = j + 8*(lane>>4).
// ---------------------------------------------------------------------------
DEV v16bf concat8(v8bf lo, v8bf hi) {
  v16bf r;
#pragma unroll
  for (int i = 0; i < 8; ++i) { r[i] = lo[i]; r[i + 8] = hi[i]; }
  return r;
}

DEV v16bf load_a_frag(const __bf16* base, int lda, int lane) {
  const int m   = lane & 15;
  const int off = (lane >> 4) * 8;
  const __bf16* p = base + (size_t)m * lda + off;
  return concat8(*(const v8bf*)p, *(const v8bf*)(p + 16));
}

DEV v16bf load_b_frag(const __bf16* base, int ldb, int lane) {
  const int n    = lane & 15;
  const int koff = (lane >> 4) * 8;
  const __bf16* p = base + (size_t)n * ldb + koff;
  return concat8(*(const v8bf*)p, *(const v8bf*)(p + 16));
}

DEV v8f wmma_bf16(v16bf a, v16bf b, v8f c) {
  return __builtin_amdgcn_wmma_f32_16x16x32_bf16(false, a, false, b, (short)0, c,
                                                 false, false);
}

DEV float rmax16(float v) {          // reduce over 16-lane half (rows stay split)
#pragma unroll
  for (int m = 1; m < 16; m <<= 1) v = fmaxf(v, __shfl_xor(v, m, 32));
  return v;
}
DEV float rsum16(float v) {
#pragma unroll
  for (int m = 1; m < 16; m <<= 1) v += __shfl_xor(v, m, 32);
  return v;
}

// Cooperative stage of a 64-row x 32-k bf16 tile (4KB) by 256 threads:
// one 16B async copy per thread. LDS layout [64][32] row-major.
DEV void stage_b_tile(const __bf16* gbase, int ld, __bf16* lbuf, int tid) {
  const int row = tid >> 2;          // 0..63
  const int kof = (tid & 3) * 8;     // 0,8,16,24
  async_copy16(gbase + (size_t)row * ld + kof, lbuf + row * 32 + kof);
}

// ---------------------------------------------------------------------------
// Stage 1: conversions / transposes to bf16 with WMMA-friendly layouts
// ---------------------------------------------------------------------------
__global__ __launch_bounds__(256) void cvt_f32_bf16(const float* __restrict__ src,
                                                    __bf16* __restrict__ dst, int n4) {
  int i = blockIdx.x * blockDim.x + threadIdx.x;
  if (i < n4) {
    float4 f = ((const float4*)src)[i];
    v4bf o = {(__bf16)f.x, (__bf16)f.y, (__bf16)f.z, (__bf16)f.w};
    ((v4bf*)dst)[i] = o;
  }
}

// W [H][D][DH] f32 -> Wt [H][DH][D] bf16
__global__ __launch_bounds__(256) void transpose_whd(const float* __restrict__ W,
                                                     __bf16* __restrict__ Wt) {
  int idx = blockIdx.x * 256 + threadIdx.x;
  if (idx < Hc * DHc * Dc) {
    int d  = idx % Dc;
    int t  = idx / Dc;
    int dh = t % DHc;
    int h  = t / DHc;
    Wt[idx] = (__bf16)W[((size_t)h * Dc + d) * DHc + dh];
  }
}

// Wo [DH*H][O] f32 -> Wot [O][DH*H] bf16
__global__ __launch_bounds__(256) void transpose_wo(const float* __restrict__ W,
                                                    __bf16* __restrict__ Wt) {
  int idx = blockIdx.x * 256 + threadIdx.x;
  if (idx < Oc * (DHc * Hc)) {
    int f = idx % (DHc * Hc);
    int o = idx / (DHc * Hc);
    Wt[idx] = (__bf16)W[(size_t)f * Oc + o];
  }
}

// ---------------------------------------------------------------------------
// Stage 2: QKV projection GEMM. Each wave: 32 rows x 64 cols (one head).
// Shared weight B-tile (64x32) staged into LDS by the whole workgroup with
// async copies, double-buffered so copy of step k+1 overlaps WMMAs of step k.
// Q pre-scaled by 1/sqrt(DH). K stored [B,H,N,DH]; V stored transposed
// [B,H,DH,N] so attention B-fragments are contiguous.
// ---------------------------------------------------------------------------
__global__ __launch_bounds__(256) void qkv_proj(
    const __bf16* __restrict__ X, const __bf16* __restrict__ Wqt,
    const __bf16* __restrict__ Wkt, const __bf16* __restrict__ Wvt,
    const float* __restrict__ bq, const float* __restrict__ bk,
    const float* __restrict__ bv, __bf16* __restrict__ Q,
    __bf16* __restrict__ K, __bf16* __restrict__ Vt) {
  __shared__ __align__(16) __bf16 Bs[2][64 * 32];

  const int mat  = blockIdx.z;                   // 0=Q 1=K 2=V
  const int h    = blockIdx.y;
  const int tid  = threadIdx.x;
  const int wave = tid >> 5;
  const int lane = tid & 31;
  const int r0   = blockIdx.x * 256 + wave * 32; // global row in [0, BN)

  const __bf16* Wt  = (mat == 0 ? Wqt : mat == 1 ? Wkt : Wvt) + (size_t)h * DHc * Dc;
  const float* bias = (mat == 0 ? bq : mat == 1 ? bk : bv) + h * DHc;

  stage_b_tile(Wt, Dc, &Bs[0][0], tid);          // preload k0 = 0

  v8f acc[2][4] = {};
  for (int k0 = 0; k0 < Dc; k0 += 32) {
    const int buf = (k0 >> 5) & 1;
    wait_async0();                               // my copies into Bs[buf] done
    __syncthreads();                             // everyone's copies visible
    if (k0 + 32 < Dc)                            // overlap next tile copy
      stage_b_tile(Wt + (k0 + 32), Dc, &Bs[buf ^ 1][0], tid);

    __builtin_prefetch(X + (size_t)r0 * Dc + k0 + 256, 0, 3);
    v16bf a0 = load_a_frag(X + (size_t)r0 * Dc + k0, Dc, lane);
    v16bf a1 = load_a_frag(X + (size_t)(r0 + 16) * Dc + k0, Dc, lane);
#pragma unroll
    for (int ci = 0; ci < 4; ++ci) {
      v16bf b = load_b_frag(&Bs[buf][(ci * 16) * 32], 32, lane);
      acc[0][ci] = wmma_bf16(a0, b, acc[0][ci]);
      acc[1][ci] = wmma_bf16(a1, b, acc[1][ci]);
    }
  }

  const int   n     = lane & 15;
  const int   mtop  = (lane >> 4) * 8;
  const float scale = (mat == 0) ? 0.125f : 1.0f;   // 1/sqrt(64) for Q
#pragma unroll
  for (int ri = 0; ri < 2; ++ri)
#pragma unroll
    for (int ci = 0; ci < 4; ++ci)
#pragma unroll
      for (int j = 0; j < 8; ++j) {
        int row  = r0 + ri * 16 + mtop + j;
        int dh   = ci * 16 + n;
        float v  = (acc[ri][ci][j] + bias[dh]) * scale;
        __bf16 o = (__bf16)v;
        int b_   = row / Nc;
        int nseq = row % Nc;
        if (mat == 2)
          Vt[(((size_t)b_ * Hc + h) * DHc + dh) * Nc + nseq] = o;
        else {
          __bf16* dst = (mat == 0) ? Q : K;
          dst[(((size_t)b_ * Hc + h) * Nc + nseq) * DHc + dh] = o;
        }
      }
}

// ---------------------------------------------------------------------------
// Stage 3: causal flash attention. One wave owns a 16-row Q tile, streams
// 32-key blocks. P transposed C-layout -> A-layout through per-wave LDS.
// Output written bf16 with feature order (dh*H + h) for the out-projection.
// ---------------------------------------------------------------------------
__global__ __launch_bounds__(256) void attn_flash(
    const __bf16* __restrict__ Q, const __bf16* __restrict__ K,
    const __bf16* __restrict__ Vt, __bf16* __restrict__ AO) {
  __shared__ __align__(16) __bf16 Pl[8][16 * 32];

  const int wave = threadIdx.x >> 5;
  const int lane = threadIdx.x & 31;
  const int h    = blockIdx.y;
  const int b    = blockIdx.z;
  const int q0   = (blockIdx.x * 8 + wave) * 16;

  const __bf16* Qb = Q  + (size_t)(b * Hc + h) * Nc * DHc;
  const __bf16* Kb = K  + (size_t)(b * Hc + h) * Nc * DHc;
  const __bf16* Vb = Vt + (size_t)(b * Hc + h) * DHc * Nc;

  const v16bf aq0 = load_a_frag(Qb + (size_t)q0 * DHc, DHc, lane);
  const v16bf aq1 = load_a_frag(Qb + (size_t)q0 * DHc + 32, DHc, lane);

  v8f   acc[4] = {};
  float mrun[8], lrun[8];
#pragma unroll
  for (int j = 0; j < 8; ++j) { mrun[j] = -1e30f; lrun[j] = 0.0f; }

  const int mtop = (lane >> 4) * 8;
  const int ncol = lane & 15;
  const int kend = q0 + 16;                       // causal: keys <= q0+15

  __bf16* pw = &Pl[wave][0];

  for (int kb = 0; kb < kend; kb += 32) {
    v8f s0 = {}, s1 = {};
    {
      v16bf b00 = load_b_frag(Kb + (size_t)kb * DHc, DHc, lane);
      v16bf b01 = load_b_frag(Kb + (size_t)kb * DHc + 32, DHc, lane);
      s0 = wmma_bf16(aq0, b00, s0);
      s0 = wmma_bf16(aq1, b01, s0);
    }
    const bool tile1 = (kb + 16 < kend);
    if (tile1) {
      v16bf b10 = load_b_frag(Kb + (size_t)(kb + 16) * DHc, DHc, lane);
      v16bf b11 = load_b_frag(Kb + (size_t)(kb + 16) * DHc + 32, DHc, lane);
      s1 = wmma_bf16(aq0, b10, s1);
      s1 = wmma_bf16(aq1, b11, s1);
    }

    // causal mask (only blocks touching the diagonal need it)
    if (kb + 31 > q0) {
#pragma unroll
      for (int j = 0; j < 8; ++j) {
        int qrow = q0 + mtop + j;
        if (kb + ncol > qrow)      s0[j] = -1e30f;
        if (kb + 16 + ncol > qrow) s1[j] = -1e30f;
      }
    }
    if (!tile1) {
#pragma unroll
      for (int j = 0; j < 8; ++j) s1[j] = -1e30f;
    }

    // online softmax update
    float alpha[8], p0[8], p1[8];
#pragma unroll
    for (int j = 0; j < 8; ++j) {
      float rm   = rmax16(fmaxf(s0[j], s1[j]));
      float newm = fmaxf(mrun[j], rm);
      alpha[j]   = __expf(mrun[j] - newm);
      mrun[j]    = newm;
      p0[j]      = __expf(s0[j] - newm);
      p1[j]      = __expf(s1[j] - newm);
      float rs   = rsum16(p0[j] + p1[j]);
      lrun[j]    = lrun[j] * alpha[j] + rs;
    }
#pragma unroll
    for (int t = 0; t < 4; ++t)
#pragma unroll
      for (int j = 0; j < 8; ++j) acc[t][j] *= alpha[j];

    // stage P (C-layout) into LDS, re-read as A-layout (in-wave transpose)
#pragma unroll
    for (int j = 0; j < 8; ++j) {
      pw[(mtop + j) * 32 + ncol]      = (__bf16)p0[j];
      pw[(mtop + j) * 32 + 16 + ncol] = (__bf16)p1[j];
    }
    asm volatile("s_wait_dscnt 0x0" ::: "memory");   // cross-lane LDS RAW
    v16bf pa = load_a_frag(pw, 32, lane);

#pragma unroll
    for (int t = 0; t < 4; ++t) {
      v16bf bv = load_b_frag(Vb + (size_t)(t * 16) * Nc + kb, Nc, lane);
      acc[t] = wmma_bf16(pa, bv, acc[t]);
    }
  }

  // normalize and store with feature order f = dh*H + h
#pragma unroll
  for (int j = 0; j < 8; ++j) lrun[j] = 1.0f / lrun[j];
#pragma unroll
  for (int t = 0; t < 4; ++t)
#pragma unroll
    for (int j = 0; j < 8; ++j) {
      int row = q0 + mtop + j;
      int dh  = t * 16 + ncol;
      AO[((size_t)(b * Nc + row)) * (DHc * Hc) + dh * Hc + h] =
          (__bf16)(acc[t][j] * lrun[j]);
    }
}

// ---------------------------------------------------------------------------
// Stage 4: output projection, f32 result + bias. Same async double-buffered
// LDS staging of the shared Wot tile as the QKV GEMM.
// ---------------------------------------------------------------------------
__global__ __launch_bounds__(256) void out_proj(
    const __bf16* __restrict__ AO, const __bf16* __restrict__ Wot,
    const float* __restrict__ bo, float* __restrict__ Y) {
  __shared__ __align__(16) __bf16 Bs[2][64 * 32];

  const int tid  = threadIdx.x;
  const int wave = tid >> 5;
  const int lane = tid & 31;
  const int r0   = blockIdx.x * 256 + wave * 32;
  const int c0   = blockIdx.y * 64;
  const int F    = DHc * Hc;                     // 1024

  const __bf16* Wtile = Wot + (size_t)c0 * F;
  stage_b_tile(Wtile, F, &Bs[0][0], tid);        // preload k0 = 0

  v8f acc[2][4] = {};
  for (int k0 = 0; k0 < F; k0 += 32) {
    const int buf = (k0 >> 5) & 1;
    wait_async0();
    __syncthreads();
    if (k0 + 32 < F)
      stage_b_tile(Wtile + (k0 + 32), F, &Bs[buf ^ 1][0], tid);

    __builtin_prefetch(AO + (size_t)r0 * F + k0 + 256, 0, 3);
    v16bf a0 = load_a_frag(AO + (size_t)r0 * F + k0, F, lane);
    v16bf a1 = load_a_frag(AO + (size_t)(r0 + 16) * F + k0, F, lane);
#pragma unroll
    for (int ci = 0; ci < 4; ++ci) {
      v16bf b = load_b_frag(&Bs[buf][(ci * 16) * 32], 32, lane);
      acc[0][ci] = wmma_bf16(a0, b, acc[0][ci]);
      acc[1][ci] = wmma_bf16(a1, b, acc[1][ci]);
    }
  }

  const int n    = lane & 15;
  const int mtop = (lane >> 4) * 8;
#pragma unroll
  for (int ri = 0; ri < 2; ++ri)
#pragma unroll
    for (int ci = 0; ci < 4; ++ci)
#pragma unroll
      for (int j = 0; j < 8; ++j) {
        int row = r0 + ri * 16 + mtop + j;
        int col = c0 + ci * 16 + n;
        Y[(size_t)row * Oc + col] = acc[ri][ci][j] + bo[col];
      }
}

// ---------------------------------------------------------------------------
extern "C" void kernel_launch(void* const* d_in, const int* in_sizes, int n_in,
                              void* d_out, int out_size, void* d_ws, size_t ws_size,
                              hipStream_t stream) {
  (void)in_sizes; (void)n_in; (void)out_size; (void)ws_size;
  const float* query = (const float*)d_in[0];
  /* d_in[1] = mask: causal tril, applied analytically from indices */
  const float* Wq = (const float*)d_in[2];
  const float* bq = (const float*)d_in[3];
  const float* Wk = (const float*)d_in[4];
  const float* bk = (const float*)d_in[5];
  const float* Wv = (const float*)d_in[6];
  const float* bv = (const float*)d_in[7];
  const float* Wo = (const float*)d_in[8];
  const float* bo = (const float*)d_in[9];
  float* out = (float*)d_out;

  // workspace layout (bf16 elements)
  __bf16* ws = (__bf16*)d_ws;
  size_t off = 0;
  __bf16* X   = ws + off; off += (size_t)BNc * Dc;          // 8M
  __bf16* Wqt = ws + off; off += (size_t)Hc * DHc * Dc;     // 1M
  __bf16* Wkt = ws + off; off += (size_t)Hc * DHc * Dc;
  __bf16* Wvt = ws + off; off += (size_t)Hc * DHc * Dc;
  __bf16* Wot = ws + off; off += (size_t)Oc * DHc * Hc;
  __bf16* Qb  = ws + off; off += (size_t)Bc * Hc * Nc * DHc;
  __bf16* Kb  = ws + off; off += (size_t)Bc * Hc * Nc * DHc;
  __bf16* Vtb = ws + off; off += (size_t)Bc * Hc * DHc * Nc;
  __bf16* AO  = ws + off; off += (size_t)BNc * DHc * Hc;

  // Stage 1: conversions
  {
    int n4 = BNc * Dc / 4;
    cvt_f32_bf16<<<(n4 + 255) / 256, 256, 0, stream>>>(query, X, n4);
    int nw = Hc * DHc * Dc;
    transpose_whd<<<(nw + 255) / 256, 256, 0, stream>>>(Wq, Wqt);
    transpose_whd<<<(nw + 255) / 256, 256, 0, stream>>>(Wk, Wkt);
    transpose_whd<<<(nw + 255) / 256, 256, 0, stream>>>(Wv, Wvt);
    int no = Oc * DHc * Hc;
    transpose_wo<<<(no + 255) / 256, 256, 0, stream>>>(Wo, Wot);
  }
  // Stage 2: QKV projection
  qkv_proj<<<dim3(BNc / 256, Hc, 3), 256, 0, stream>>>(X, Wqt, Wkt, Wvt, bq, bk,
                                                       bv, Qb, Kb, Vtb);
  // Stage 3: flash attention (causal)
  attn_flash<<<dim3(Nc / 16 / 8, Hc, Bc), 256, 0, stream>>>(Qb, Kb, Vtb, AO);
  // Stage 4: output projection
  out_proj<<<dim3(BNc / 256, Oc / 64), 256, 0, stream>>>(AO, Wot, bo, out);
}
